// SoftmaxEncoder_20160576488053
// MI455X (gfx1250) — matrix operand order repaired and checked by reading, not compiled, not run on hardware
//
#include <hip/hip_runtime.h>
#include <hip/hip_bf16.h>

#define DEVI __device__ __forceinline__

typedef __attribute__((ext_vector_type(16))) __bf16 bfrag_t;  // WMMA A/B fragment
typedef __attribute__((ext_vector_type(8)))  __bf16 bvec8;    // 16-byte bf16 vector
typedef __attribute__((ext_vector_type(4)))  __bf16 bvec4;    // 8-byte bf16 vector
typedef __attribute__((ext_vector_type(8)))  float  afrag_t;  // WMMA C/D fragment
typedef __attribute__((ext_vector_type(4)))  float  f4;

constexpr int BATCH  = 64;
constexpr int DDIM   = 32;
constexpr int EDIM   = 128;
constexpr int NHEAD  = 8;
constexpr int NLAYER = 12;
constexpr int SMAX   = 448;
constexpr int SBASE  = 288;   // N + d
constexpr int NCOT   = 5;

// ---------------- WMMA fragment helpers (vectorized, CDNA5 wave32 layouts) ----------------

// A-matrix 16x32 (MxK) from row-major source (stride lda elems, 16B-aligned rows).
DEVI bfrag_t load_a_frag(const __bf16* a, int lda) {
  const int lane = threadIdx.x & 31;
  const int m    = lane & 15;
  const int half = lane >> 4;
  const __bf16* p = a + m * lda + half * 8;
  bvec8 lo = *(const bvec8*)(p);
  bvec8 hi = *(const bvec8*)(p + 16);
  return __builtin_shufflevector(lo, hi, 0, 1, 2, 3, 4, 5, 6, 7,
                                 8, 9, 10, 11, 12, 13, 14, 15);
}

// B-matrix 32x16 (KxN) where logical B[k][n] = src[n*ldt + k] (source is N x K row-major).
DEVI bfrag_t load_bT_frag(const __bf16* src, int ldt) {
  const int lane = threadIdx.x & 31;
  const int n    = lane & 15;
  const int half = lane >> 4;
  const __bf16* p = src + n * ldt + half * 8;
  bvec8 lo = *(const bvec8*)(p);
  bvec8 hi = *(const bvec8*)(p + 16);
  return __builtin_shufflevector(lo, hi, 0, 1, 2, 3, 4, 5, 6, 7,
                                 8, 9, 10, 11, 12, 13, 14, 15);
}

DEVI afrag_t wmma_bf16(bfrag_t a, bfrag_t b, afrag_t c) {
  return __builtin_amdgcn_wmma_f32_16x16x32_bf16(false, a, false, b, (short)0, c, false, false);
}

// ---------------- weight conversion (fp32 -> bf16, pre-transposed for bT loads) ----------

// dst[(l, h, c, j)] = src[(l, j, h*128 + c)]
__global__ void k_cvt_qkvT(const float* __restrict__ src, __bf16* __restrict__ dst, int n) {
  int i = blockIdx.x * blockDim.x + threadIdx.x;
  if (i >= n) return;
  const int j = i & 127;
  const int c = (i >> 7) & 127;
  const int h = (i >> 14) & 7;
  const int l = i >> 17;
  dst[i] = (__bf16)src[((size_t)(l * EDIM + j)) * (NHEAD * EDIM) + h * EDIM + c];
}

// dst[(l, c, j)] = src[(l, j, c)]
__global__ void k_cvt_mlpT(const float* __restrict__ src, __bf16* __restrict__ dst, int n) {
  int i = blockIdx.x * blockDim.x + threadIdx.x;
  if (i >= n) return;
  const int j = i & 127;
  const int c = (i >> 7) & 127;
  const int l = i >> 14;
  dst[i] = (__bf16)src[((size_t)(l * EDIM + j)) * EDIM + c];
}

// ---------------- setup kernels ----------------

__global__ void k_init(const float* __restrict__ xs, const float* __restrict__ ys,
                       const float* __restrict__ Win, const float* __restrict__ bin,
                       float* __restrict__ Hf) {
  const int b  = blockIdx.x;
  const int rt = blockIdx.y;
  for (int idx = threadIdx.x; idx < 16 * EDIM; idx += blockDim.x) {
    const int r = rt * 16 + (idx >> 7);
    const int e = idx & 127;
    float acc = bin[e];
    if (r < 256) {
      const float* y = ys + ((size_t)b * 256 + r) * DDIM;
      for (int j = 0; j < DDIM; ++j) acc += y[j] * Win[j * EDIM + e];
      if (r < 5) {  // HALF_N
        const float* x = xs + ((size_t)b * 256 + r) * DDIM;
        for (int j = 0; j < DDIM; ++j) acc += x[j] * Win[(DDIM + j) * EDIM + e];
      }
    } else {
      acc += Win[(2 * DDIM + (r - 256)) * EDIM + e];
    }
    Hf[((size_t)b * SMAX + r) * EDIM + e] = acc;
  }
}

// Hb = bf16(Hf); Af = Hf (vectorized: b128 loads, packed bf16 stores)
__global__ void k_prep(const float* __restrict__ Hf, __bf16* __restrict__ Hb,
                       float* __restrict__ Af, int S) {
  const int b = blockIdx.x;
  const f4* h = (const f4*)(Hf + (size_t)b * SMAX * EDIM);
  bvec4*   hb = (bvec4*)(Hb + (size_t)b * SMAX * EDIM);
  f4*      af = (f4*)(Af + (size_t)b * SMAX * EDIM);
  const int nq = S * EDIM / 4;
  for (int i = blockIdx.y * blockDim.x + threadIdx.x; i < nq; i += gridDim.y * blockDim.x) {
    const f4 v = h[i];
    af[i] = v;
    bvec4 bv;
#pragma unroll
    for (int j = 0; j < 4; ++j) bv[j] = (__bf16)v[j];
    hb[i] = bv;
  }
}

// ------- per-head K/V projection: 32 rows/block, B-fragments shared across row pair -----

__global__ void k_kvproj(const __bf16* __restrict__ Hb,
                         const __bf16* __restrict__ WkT, const __bf16* __restrict__ WvT,
                         __bf16* __restrict__ Kb, __bf16* __restrict__ VbT,
                         int S, int l, int h) {
  const int b    = blockIdx.x;
  const int row0 = blockIdx.y * 32;
  const int wave = threadIdx.x >> 5;

  const __bf16* hb0 = Hb + ((size_t)b * SMAX + row0) * EDIM;
  const __bf16* hb1 = hb0 + 16 * EDIM;
  const size_t wb   = (((size_t)l * NHEAD + h) * EDIM + wave * 16) * EDIM;

  __builtin_prefetch(WkT + wb, 0, 3);
  __builtin_prefetch(WvT + wb, 0, 3);

  afrag_t accK0 = {}, accK1 = {}, accV0 = {}, accV1 = {};
#pragma unroll
  for (int kk = 0; kk < EDIM; kk += 32) {
    bfrag_t a0 = load_a_frag(hb0 + kk, EDIM);
    bfrag_t a1 = load_a_frag(hb1 + kk, EDIM);
    bfrag_t bk = load_bT_frag(WkT + wb + kk, EDIM);
    accK0 = wmma_bf16(a0, bk, accK0);
    accK1 = wmma_bf16(a1, bk, accK1);
    bfrag_t bv = load_bT_frag(WvT + wb + kk, EDIM);
    accV0 = wmma_bf16(a0, bv, accV0);
    accV1 = wmma_bf16(a1, bv, accV1);
  }

  const int lane = threadIdx.x & 31;
  const int n = lane & 15, half = lane >> 4;

  // K stored row-major (read contiguously by the Q.K^T loop)
  __bf16* ko = Kb + ((size_t)b * SMAX + row0) * EDIM + wave * 16 + n;
#pragma unroll
  for (int r = 0; r < 8; ++r) {
    ko[(size_t)(r + half * 8) * EDIM]      = (__bf16)accK0[r];
    ko[(size_t)(16 + r + half * 8) * EDIM] = (__bf16)accK1[r];
  }

  // V stored transposed [col][row], one packed 16B store per lane per tile
  __bf16* vo = VbT + ((size_t)b * EDIM + wave * 16 + n) * SMAX + row0 + half * 8;
  bvec8 pv0, pv1;
#pragma unroll
  for (int r = 0; r < 8; ++r) { pv0[r] = (__bf16)accV0[r]; pv1[r] = (__bf16)accV1[r]; }
  *(bvec8*)(vo)      = pv0;
  *(bvec8*)(vo + 16) = pv1;
}

// ------- per-head attention: 32 query rows/block, Af += softmax(Q K^T) V ---------------

__global__ void k_attn(const __bf16* __restrict__ Hb, const __bf16* __restrict__ WqT,
                       const __bf16* __restrict__ Kb, const __bf16* __restrict__ VbT,
                       float* __restrict__ Af, int S, int l, int h) {
  extern __shared__ char smem[];
  __bf16* sQ = (__bf16*)smem;               // 32 x 128   (8 KB)
  float*  sS = (float*)(sQ + 32 * EDIM);    // 32 x SMAX  (56 KB)
  __bf16* sP = (__bf16*)(sS + 32 * SMAX);   // 32 x SMAX  (28 KB)

  const int b    = blockIdx.x;
  const int row0 = blockIdx.y * 32;
  const int wave = threadIdx.x >> 5;
  const int lane = threadIdx.x & 31;
  const int n = lane & 15, half = lane >> 4;

  const __bf16* hb0 = Hb + ((size_t)b * SMAX + row0) * EDIM;
  const __bf16* hb1 = hb0 + 16 * EDIM;
  const size_t wb   = (((size_t)l * NHEAD + h) * EDIM + wave * 16) * EDIM;
  __builtin_prefetch(WqT + wb, 0, 3);

  // Q tile (32x128): each wave computes one 16-wide col tile for both row tiles
  {
    afrag_t acc0 = {}, acc1 = {};
#pragma unroll
    for (int kk = 0; kk < EDIM; kk += 32) {
      bfrag_t w  = load_bT_frag(WqT + wb + kk, EDIM);
      bfrag_t a0 = load_a_frag(hb0 + kk, EDIM);
      acc0 = wmma_bf16(a0, w, acc0);
      bfrag_t a1 = load_a_frag(hb1 + kk, EDIM);
      acc1 = wmma_bf16(a1, w, acc1);
    }
#pragma unroll
    for (int r = 0; r < 8; ++r) {
      sQ[(r + half * 8) * EDIM + wave * 16 + n]        = (__bf16)acc0[r];
      sQ[(16 + r + half * 8) * EDIM + wave * 16 + n]   = (__bf16)acc1[r];
    }
  }
  __syncthreads();

  // scores = Q @ K^T ; hoist Q A-fragments (both row tiles), share K frags across rows
  {
    bfrag_t qa0[4], qa1[4];
#pragma unroll
    for (int t = 0; t < 4; ++t) {
      qa0[t] = load_a_frag(sQ + t * 32, EDIM);
      qa1[t] = load_a_frag(sQ + 16 * EDIM + t * 32, EDIM);
    }
    const int nkt = S / 16;
    for (int j = wave; j < nkt; j += 8) {
      const __bf16* kb = Kb + ((size_t)b * SMAX + j * 16) * EDIM;
      afrag_t acc0 = {}, acc1 = {};
#pragma unroll
      for (int t = 0; t < 4; ++t) {
        bfrag_t bt = load_bT_frag(kb + t * 32, EDIM);
        acc0 = wmma_bf16(qa0[t], bt, acc0);
        acc1 = wmma_bf16(qa1[t], bt, acc1);
      }
#pragma unroll
      for (int r = 0; r < 8; ++r) {
        sS[(r + half * 8) * SMAX + j * 16 + n]      = acc0[r];
        sS[(16 + r + half * 8) * SMAX + j * 16 + n] = acc1[r];
      }
    }
  }
  __syncthreads();

  // softmax over 32 rows (8 threads/row, shfl_xor width-8 reduction)
  {
    const int r  = threadIdx.x >> 3;
    const int c0 = threadIdx.x & 7;
    float mx = -1e30f;
    for (int c = c0; c < S; c += 8) mx = fmaxf(mx, sS[r * SMAX + c]);
    for (int o = 4; o >= 1; o >>= 1) mx = fmaxf(mx, __shfl_xor(mx, o, 8));
    float sum = 0.f;
    for (int c = c0; c < S; c += 8) {
      const float ev = __expf(sS[r * SMAX + c] - mx);
      sS[r * SMAX + c] = ev;
      sum += ev;
    }
    for (int o = 4; o >= 1; o >>= 1) sum += __shfl_xor(sum, o, 8);
    const float inv = 1.f / sum;
    for (int c = c0; c < S; c += 8) sP[r * SMAX + c] = (__bf16)(sS[r * SMAX + c] * inv);
  }
  __syncthreads();

  // O = P @ V (V pre-transposed); share V fragments across both row tiles
  {
    const __bf16* vt = VbT + ((size_t)b * EDIM + wave * 16) * SMAX;
    afrag_t acc0 = {}, acc1 = {};
    for (int kk = 0; kk < S; kk += 32) {
      bfrag_t bv = load_bT_frag(vt + kk, SMAX);
      bfrag_t a0 = load_a_frag(sP + kk, SMAX);
      acc0 = wmma_bf16(a0, bv, acc0);
      bfrag_t a1 = load_a_frag(sP + 16 * SMAX + kk, SMAX);
      acc1 = wmma_bf16(a1, bv, acc1);
    }
    float* ao = Af + ((size_t)b * SMAX + row0) * EDIM + wave * 16 + n;
#pragma unroll
    for (int r = 0; r < 8; ++r) {
      ao[(size_t)(r + half * 8) * EDIM]      += acc0[r];
      ao[(size_t)(16 + r + half * 8) * EDIM] += acc1[r];
    }
  }
}

// ---------------- LayerNorm: Hf = LN(Af)*g + b ; Hb = bf16(Hf) ----------------

__global__ void k_ln(const float* __restrict__ Af, const float* __restrict__ g,
                     const float* __restrict__ bb, float* __restrict__ Hf,
                     __bf16* __restrict__ Hb, int l) {
  __shared__ float red1[4], red2[4];
  const int b = blockIdx.x, s = blockIdx.y, t = threadIdx.x;  // 128 threads
  const size_t idx = ((size_t)b * SMAX + s) * EDIM + t;
  const float v = Af[idx];
  float sum = v;
  for (int o = 16; o >= 1; o >>= 1) sum += __shfl_xor(sum, o, 32);
  if ((t & 31) == 0) red1[t >> 5] = sum;
  __syncthreads();
  const float mu = (red1[0] + red1[1] + red1[2] + red1[3]) * (1.f / 128.f);
  const float dv = v - mu;
  float sq = dv * dv;
  for (int o = 16; o >= 1; o >>= 1) sq += __shfl_xor(sq, o, 32);
  if ((t & 31) == 0) red2[t >> 5] = sq;
  __syncthreads();
  const float var = (red2[0] + red2[1] + red2[2] + red2[3]) * (1.f / 128.f);
  const float y = dv * rsqrtf(var + 1e-5f) * g[l * EDIM + t] + bb[l * EDIM + t];
  Hf[idx] = y;
  Hb[idx] = (__bf16)y;
}

// ---------------- MLP: 32 rows/block, weight fragments shared across row pair ----------

__global__ void k_mlp(const __bf16* __restrict__ Hb, const __bf16* __restrict__ W1T,
                      const float* __restrict__ b1, const __bf16* __restrict__ W2T,
                      const float* __restrict__ b2, float* __restrict__ Hf, int l) {
  extern __shared__ char smem[];
  __bf16* sT = (__bf16*)smem;  // 32 x 128 hidden tile (8 KB)

  const int b    = blockIdx.x;
  const int row0 = blockIdx.y * 32;
  const int wave = threadIdx.x >> 5;
  const int lane = threadIdx.x & 31;
  const int n = lane & 15, half = lane >> 4;

  const __bf16* hb0 = Hb + ((size_t)b * SMAX + row0) * EDIM;
  const __bf16* hb1 = hb0 + 16 * EDIM;
  const size_t wb   = ((size_t)l * EDIM + wave * 16) * EDIM;
  __builtin_prefetch(W1T + wb, 0, 3);
  __builtin_prefetch(W2T + wb, 0, 3);

  // T = relu(H@W1 + b1)
  {
    afrag_t acc0 = {}, acc1 = {};
#pragma unroll
    for (int kk = 0; kk < EDIM; kk += 32) {
      bfrag_t w  = load_bT_frag(W1T + wb + kk, EDIM);
      bfrag_t a0 = load_a_frag(hb0 + kk, EDIM);
      acc0 = wmma_bf16(a0, w, acc0);
      bfrag_t a1 = load_a_frag(hb1 + kk, EDIM);
      acc1 = wmma_bf16(a1, w, acc1);
    }
    const float bv = b1[l * EDIM + wave * 16 + n];
#pragma unroll
    for (int r = 0; r < 8; ++r) {
      float t0 = acc0[r] + bv; t0 = t0 > 0.f ? t0 : 0.f;
      float t1 = acc1[r] + bv; t1 = t1 > 0.f ? t1 : 0.f;
      sT[(r + half * 8) * EDIM + wave * 16 + n]      = (__bf16)t0;
      sT[(16 + r + half * 8) * EDIM + wave * 16 + n] = (__bf16)t1;
    }
  }
  __syncthreads();

  // Hf += T@W2 + b2
  {
    afrag_t acc0 = {}, acc1 = {};
#pragma unroll
    for (int kk = 0; kk < EDIM; kk += 32) {
      bfrag_t w  = load_bT_frag(W2T + wb + kk, EDIM);
      bfrag_t a0 = load_a_frag(sT + kk, EDIM);
      acc0 = wmma_bf16(a0, w, acc0);
      bfrag_t a1 = load_a_frag(sT + 16 * EDIM + kk, EDIM);
      acc1 = wmma_bf16(a1, w, acc1);
    }
    const float bv = b2[l * EDIM + wave * 16 + n];
    float* ho = Hf + ((size_t)b * SMAX + row0) * EDIM + wave * 16 + n;
#pragma unroll
    for (int r = 0; r < 8; ++r) {
      ho[(size_t)(r + half * 8) * EDIM]      += acc0[r] + bv;
      ho[(size_t)(16 + r + half * 8) * EDIM] += acc1[r] + bv;
    }
  }
}

// ---------------- CoT append + readout ----------------

__global__ void k_append(const float* __restrict__ Hf, float* __restrict__ Hs, int Sin) {
  const int b = blockIdx.x;
  for (int i = threadIdx.x; i < DDIM * EDIM; i += blockDim.x) {
    const int r = i >> 7, c = i & 127;
    float v = Hf[((size_t)b * SMAX + (Sin - DDIM) + r) * EDIM + c];
    if (c >= 2 * DDIM && c < 3 * DDIM) v = (c - 2 * DDIM == r) ? 1.f : 0.f;
    Hs[((size_t)b * SMAX + Sin + r) * EDIM + c] = v;
  }
}

__global__ void k_readout(const float* __restrict__ Hf, const float* __restrict__ Wout,
                          const float* __restrict__ bout, float* __restrict__ out) {
  const int b = blockIdx.x;
  for (int i = threadIdx.x; i < 11 * 96; i += blockDim.x) {
    const int p = i / 96, c = i % 96;
    float acc = bout[DDIM + c];
    const float* h = Hf + ((size_t)b * SMAX + p) * EDIM;
    for (int j = 0; j < EDIM; ++j) acc += h[j] * Wout[j * EDIM + DDIM + c];
    out[((size_t)b * 11 + p) * 96 + c] = acc;
  }
}

// ---------------- host orchestration ----------------

extern "C" void kernel_launch(void* const* d_in, const int* in_sizes, int n_in,
                              void* d_out, int out_size, void* d_ws, size_t ws_size,
                              hipStream_t stream) {
  (void)in_sizes; (void)n_in; (void)out_size; (void)ws_size;
  const float* xs   = (const float*)d_in[0];
  const float* ys   = (const float*)d_in[1];
  /* d_in[2] head_mask: unused by the reference computation */
  const float* win  = (const float*)d_in[3];
  const float* bin  = (const float*)d_in[4];
  const float* wq   = (const float*)d_in[5];
  const float* wk   = (const float*)d_in[6];
  const float* wv   = (const float*)d_in[7];
  const float* lng  = (const float*)d_in[8];
  const float* lnb  = (const float*)d_in[9];
  const float* w1   = (const float*)d_in[10];
  const float* b1   = (const float*)d_in[11];
  const float* w2   = (const float*)d_in[12];
  const float* b2   = (const float*)d_in[13];
  const float* wout = (const float*)d_in[14];
  const float* bout = (const float*)d_in[15];
  float* out = (float*)d_out;

  char* ws = (char*)d_ws;
  size_t off = 0;
  auto carve = [&](size_t bytes) -> char* {
    char* p = ws + off;
    off = (off + bytes + 255) & ~(size_t)255;
    return p;
  };
  const size_t actF = (size_t)BATCH * SMAX * EDIM;
  float*  Hf  = (float*)carve(actF * 4);
  float*  Hs  = (float*)carve(actF * 4);
  float*  Af  = (float*)carve(actF * 4);
  __bf16* Hb  = (__bf16*)carve(actF * 2);
  __bf16* Kb  = (__bf16*)carve(actF * 2);
  __bf16* VbT = (__bf16*)carve(actF * 2);
  const int NQKV = NLAYER * NHEAD * EDIM * EDIM;
  const int NMLP = NLAYER * EDIM * EDIM;
  __bf16* WqT = (__bf16*)carve((size_t)NQKV * 2);
  __bf16* WkT = (__bf16*)carve((size_t)NQKV * 2);
  __bf16* WvT = (__bf16*)carve((size_t)NQKV * 2);
  __bf16* W1T = (__bf16*)carve((size_t)NMLP * 2);
  __bf16* W2T = (__bf16*)carve((size_t)NMLP * 2);

  k_cvt_qkvT<<<(NQKV + 255) / 256, 256, 0, stream>>>(wq, WqT, NQKV);
  k_cvt_qkvT<<<(NQKV + 255) / 256, 256, 0, stream>>>(wk, WkT, NQKV);
  k_cvt_qkvT<<<(NQKV + 255) / 256, 256, 0, stream>>>(wv, WvT, NQKV);
  k_cvt_mlpT<<<(NMLP + 255) / 256, 256, 0, stream>>>(w1, W1T, NMLP);
  k_cvt_mlpT<<<(NMLP + 255) / 256, 256, 0, stream>>>(w2, W2T, NMLP);

  k_init<<<dim3(BATCH, SBASE / 16), 256, 0, stream>>>(xs, ys, win, bin, Hf);

  constexpr size_t ATTN_SMEM = 32 * EDIM * sizeof(__bf16)
                             + 32 * SMAX * sizeof(float) + 32 * SMAX * sizeof(__bf16);
  constexpr size_t MLP_SMEM  = 32 * EDIM * sizeof(__bf16);

  for (int cot = 0; cot < NCOT; ++cot) {
    const int S = SBASE + DDIM * cot;
    hipMemcpyAsync(Hs, Hf, actF * 4, hipMemcpyDeviceToDevice, stream);  // save H_input

    for (int l = 0; l < NLAYER; ++l) {
      k_prep<<<dim3(BATCH, 32), 256, 0, stream>>>(Hf, Hb, Af, S);
      for (int h = 0; h < NHEAD; ++h) {
        k_kvproj<<<dim3(BATCH, S / 32), 256, 0, stream>>>(Hb, WkT, WvT, Kb, VbT, S, l, h);
        k_attn<<<dim3(BATCH, S / 32), 256, ATTN_SMEM, stream>>>(Hb, WqT, Kb, VbT, Af, S, l, h);
      }
      k_ln<<<dim3(BATCH, S), 128, 0, stream>>>(Af, lng, lnb, Hf, Hb, l);
      k_mlp<<<dim3(BATCH, S / 32), 256, MLP_SMEM, stream>>>(Hb, W1T, b1, W2T, b2, Hf, l);
    }

    k_append<<<BATCH, 128, 0, stream>>>(Hf, Hs, S);
    hipMemcpyAsync(Hf, Hs, actF * 4, hipMemcpyDeviceToDevice, stream);  // restore + append
  }

  k_readout<<<BATCH, 128, 0, stream>>>(Hf, wout, bout, out);
}
